// S4Layer_53300544143477
// MI455X (gfx1250) — compile-verified, matrix operand order checked by reading, not run on hardware
//
#include <hip/hip_runtime.h>

// S4 layer forward for MI455X (gfx1250, wave32, WMMA + TDM).
// Pipeline:
//   1) s4_cauchy_kernel     : VALU f32 Cauchy resolvent -> at_roots f16 [Re|Im] rows
//   2) s4_twiddle_kernel    : unit-scale iDFT matrix, transposed f16 [j][4096]
//   3) s4_transpose_u_kernel: u (b,l,h) f32 -> Ut2 [h][16][l] f16 (batch 8->16 zero pad)
//   4) s4_krev_pad_kernel   : zero tail padding of Krev rows
//   5) s4_idft_wmma_kernel  : Krev = reverse((1/L) * A16 @ W)    (v_wmma_f32_16x16x32_f16)
//   6) s4_conv_wmma_kernel  : Y = ToeplitzLower(K_h) @ u_h + D*u (TDM -> LDS, WMMA)

#define H_  512
#define N_  64
#define L_  2048
#define B_  8
#define K2_ 4096       // [Re|Im] concatenated reduction dim for the iDFT GEMM
#define KREV_LD 2112   // 2048 + 64 zero-pad tail (covers Toeplitz overhang <= 46)

typedef __attribute__((ext_vector_type(16))) _Float16 v16h;
typedef __attribute__((ext_vector_type(8)))  float    v8f;
typedef __attribute__((ext_vector_type(4)))  unsigned uint4v;
typedef __attribute__((ext_vector_type(8)))  int      int8v;
typedef __attribute__((ext_vector_type(4)))  int      int4v;

// Load one lane's 16 f16 elements of a 16x32 A-tile (or B^T-tile) from a
// row-major matrix. Per ISA 7.12.2 (16-bit A 16x32): lane half 0 holds
// K = {k0..k0+7, k0+16..k0+23}, half 1 holds K = {k0+8..k0+15, k0+24..k0+31}.
static __device__ __forceinline__ v16h load_tile_rowmajor(
    const _Float16* __restrict__ row, int k0, int half) {
  v16h v;
  const int b1 = k0 + 8 * half;
  const int b2 = k0 + 16 + 8 * half;
#pragma unroll
  for (int i = 0; i < 8; ++i) v[i] = row[b1 + i];
#pragma unroll
  for (int i = 0; i < 8; ++i) v[8 + i] = row[b2 + i];
  return v;
}

// ---------------------------------------------------------------------------
// Stage 1: Cauchy resolvent. One block = 256 l-values of one h.
// ---------------------------------------------------------------------------
__global__ void s4_cauchy_kernel(const float* __restrict__ LRe,
                                 const float* __restrict__ LIm,
                                 const float* __restrict__ P,
                                 const float* __restrict__ Bm,
                                 const float* __restrict__ Cm,
                                 const float* __restrict__ logstep,
                                 _Float16* __restrict__ A16) {
  __shared__ float lr[N_], li[N_];
  __shared__ float w00r[N_], w00i[N_], w01r[N_], w01i[N_];
  __shared__ float w10r[N_], w10i[N_], w11r[N_];

  const int h = blockIdx.x >> 3;
  const int l = ((blockIdx.x & 7) << 8) + threadIdx.x;

  const int n = threadIdx.x;
  if (n < N_) {
    const int base = (h * N_ + n) * 2;
    const float pr = P[base],  pi = P[base + 1];
    const float br = Bm[base], bi = Bm[base + 1];
    const float cr = Cm[base], ci = Cm[base + 1];
    // a0 = conj(C), a1 = conj(P), b0 = B, b1 = P
    w00r[n] = cr * br + ci * bi;  w00i[n] = cr * bi - ci * br;
    w01r[n] = cr * pr + ci * pi;  w01i[n] = cr * pi - ci * pr;
    w10r[n] = pr * br + pi * bi;  w10i[n] = pr * bi - pi * br;
    w11r[n] = pr * pr + pi * pi;
    lr[n] = fminf(LRe[h * N_ + n], -1e-4f);
    li[n] = LIm[h * N_ + n];
  }
  __syncthreads();

  const float step = __expf(logstep[h]);

  const float th = -6.28318530717958647692f * (float)l * (1.0f / (float)L_);
  float s, c;
  __sincosf(th, &s, &c);
  const float wre = c, wim = s;

  const float dre = 1.0f + wre, dim = wim;
  const float inv_den = 1.0f / (dre * dre + dim * dim);
  const float csr = 2.0f * dre * inv_den;
  const float csi = -2.0f * dim * inv_den;
  const float nre = 1.0f - wre, nim = -wim;
  const float tre = (nre * dre + nim * dim) * inv_den;
  const float tim = (nim * dre - nre * dim) * inv_den;
  const float gs  = 2.0f / step;
  const float gre = gs * tre, gim = gs * tim;

  float k00r = 0.f, k00i = 0.f, k01r = 0.f, k01i = 0.f;
  float k10r = 0.f, k10i = 0.f, k11r = 0.f, k11i = 0.f;
#pragma unroll 8
  for (int nn = 0; nn < N_; ++nn) {
    const float dr = gre - lr[nn];
    const float di = gim - li[nn];
    const float ir = 1.0f / (dr * dr + di * di);
    const float rre = dr * ir, rim = -di * ir;
    k00r += w00r[nn] * rre - w00i[nn] * rim;
    k00i += w00r[nn] * rim + w00i[nn] * rre;
    k01r += w01r[nn] * rre - w01i[nn] * rim;
    k01i += w01r[nn] * rim + w01i[nn] * rre;
    k10r += w10r[nn] * rre - w10i[nn] * rim;
    k10i += w10r[nn] * rim + w10i[nn] * rre;
    k11r += w11r[nn] * rre;
    k11i += w11r[nn] * rim;
  }

  const float er = 1.0f + k11r, ei = k11i;
  const float ed = 1.0f / (er * er + ei * ei);
  const float numr = k01r * k10r - k01i * k10i;
  const float numi = k01r * k10i + k01i * k10r;
  const float qr = (numr * er + numi * ei) * ed;
  const float qi = (numi * er - numr * ei) * ed;

  const float ar = k00r - qr, ai = k00i - qi;
  const float atr = csr * ar - csi * ai;
  const float ati = csr * ai + csi * ar;

  A16[(size_t)h * K2_ + l]      = (_Float16)atr;
  A16[(size_t)h * K2_ + L_ + l] = (_Float16)ati;
}

// ---------------------------------------------------------------------------
// Stage 2a: transposed iDFT twiddle matrix Wt[j][lp] (unit scale).
// ---------------------------------------------------------------------------
__global__ void s4_twiddle_kernel(_Float16* __restrict__ Wt) {
  const int gid = blockIdx.x * 256 + threadIdx.x;
  const int j  = gid >> 12;
  const int lp = gid & (K2_ - 1);
  const int l  = lp & (L_ - 1);
  const int ph = (l * j) & (L_ - 1);
  const float th = 6.28318530717958647692f * (1.0f / (float)L_) * (float)ph;
  float s, c;
  __sincosf(th, &s, &c);
  const float val = (lp < L_) ? c : -s;
  Wt[(size_t)j * K2_ + lp] = (_Float16)val;
}

// ---------------------------------------------------------------------------
// Stage 2b: u (b,l,h) f32 -> Ut2 [h][16][l] f16, batch rows 8..15 zeroed.
// ---------------------------------------------------------------------------
__global__ void s4_transpose_u_kernel(const float* __restrict__ u,
                                      _Float16* __restrict__ Ut2) {
  const int gid = blockIdx.x * 256 + threadIdx.x;   // gid = h*L + l
  const int h = gid >> 11;
  const int l = gid & (L_ - 1);
  _Float16* w = Ut2 + (size_t)h * 16 * L_ + l;
#pragma unroll
  for (int b = 0; b < B_; ++b)
    w[(size_t)b * L_] = (_Float16)u[((size_t)b * L_ + l) * H_ + h];
#pragma unroll
  for (int b = B_; b < 16; ++b) w[(size_t)b * L_] = (_Float16)0.f;
}

// ---------------------------------------------------------------------------
// Stage 2c: zero the 64-element tail of each Krev row (Toeplitz overhang pad).
// ---------------------------------------------------------------------------
__global__ void s4_krev_pad_kernel(_Float16* __restrict__ Krev) {
  const int gid = blockIdx.x * 256 + threadIdx.x;   // H_*64 threads
  const int h = gid >> 6;
  const int p = gid & 63;
  Krev[(size_t)h * KREV_LD + L_ + p] = (_Float16)0.f;
}

// ---------------------------------------------------------------------------
// Stage 3: Krev(512 x 2112) = reverse_j((1/L) * A16 @ W), f16 WMMA.
// Krev[h][p] = K[h][2047-p]; one wave per 16x16 tile, 8 waves/block.
// ---------------------------------------------------------------------------
__global__ void s4_idft_wmma_kernel(const _Float16* __restrict__ A16,
                                    const _Float16* __restrict__ Wt,
                                    _Float16* __restrict__ Krev) {
  const int lane = threadIdx.x & 31;
  const int tile = blockIdx.x * 8 + (threadIdx.x >> 5);
  const int mt = tile >> 7;   // h tile
  const int nt = tile & 127;  // j tile
  const int r = lane & 15, half = lane >> 4;

  const _Float16* __restrict__ arow = A16 + (size_t)(mt * 16 + r) * K2_;
  const _Float16* __restrict__ brow = Wt  + (size_t)(nt * 16 + r) * K2_;

  v8f acc = {};
  for (int k0 = 0; k0 < K2_; k0 += 32) {
    __builtin_prefetch(arow + k0 + 128, 0, 1);  // global_prefetch_b8
    __builtin_prefetch(brow + k0 + 128, 0, 1);
    const v16h a = load_tile_rowmajor(arow, k0, half);
    const v16h b = load_tile_rowmajor(brow, k0, half);
    acc = __builtin_amdgcn_wmma_f32_16x16x32_f16(false, a, false, b,
                                                 (short)0, acc, false, false);
  }
#pragma unroll
  for (int i = 0; i < 8; ++i) {
    const int m = mt * 16 + i + 8 * half;   // C/D: M = i + 8*half
    const int j = nt * 16 + r;              //      N = lane & 15
    Krev[(size_t)m * KREV_LD + (L_ - 1 - j)] =
        (_Float16)(acc[i] * (1.0f / (float)L_));
  }
}

// ---------------------------------------------------------------------------
// Stage 4: causal conv as lower-triangular Toeplitz GEMM per h.
// Block = 8 consecutive l-tiles of one h. The full u-slab for h
// (Ut2[h] : 16 x 2048 f16 = 64 KB) is DMA'd to LDS once via the
// Tensor Data Mover; B tiles then come from LDS, A tiles from Krev
// (forward-contiguous thanks to the reversed layout).
// ---------------------------------------------------------------------------
__global__ void s4_conv_wmma_kernel(const _Float16* __restrict__ Krev,
                                    const _Float16* __restrict__ Ut2,
                                    const float* __restrict__ u,
                                    const float* __restrict__ D,
                                    float* __restrict__ out) {
  __shared__ _Float16 ldsU[16 * L_];   // 64 KB

  const int lane = threadIdx.x & 31;
  const int tile = blockIdx.x * 8 + (threadIdx.x >> 5);
  const int h  = tile >> 7;    // all 8 waves in a block share h
  const int lt = tile & 127;
  const int l0 = lt * 16;
  const int r = lane & 15, half = lane >> 4;

  // --- TDM: load Ut2[h] (2D tile 2048 x 16, 2-byte elements) into LDS ------
  if (threadIdx.x < 32) {  // one wave issues the descriptor (EXEC-uniform)
    const unsigned long long ga =
        (unsigned long long)(uintptr_t)(Ut2 + (size_t)h * 16 * L_);
    const unsigned lds_off = (unsigned)(uintptr_t)(&ldsU[0]);
    uint4v g0;
    g0[0] = 1u;                                   // count=1, user descriptor
    g0[1] = lds_off;                              // lds_addr (bytes)
    g0[2] = (unsigned)(ga & 0xFFFFFFFFu);         // global_addr[31:0]
    g0[3] = (unsigned)((ga >> 32) & 0x1FFFFFFu)   // global_addr[56:32]
            | 0x80000000u;                        // type=2 (bits 127:126)
    int8v g1;
    g1[0] = 0x00010000;          // data_size=1 (2B), workgroup_mask=0
    g1[1] = (int)(2048u << 16);  // tensor_dim0[15:0]=2048 (bits 63:48)
    g1[2] = (int)(16u << 16);    // tensor_dim0 hi=0 | tensor_dim1[15:0]=16
    g1[3] = (int)(2048u << 16);  // tensor_dim1 hi=0 | tile_dim0=2048
    g1[4] = 16;                  // tile_dim1=16, tile_dim2=0
    g1[5] = 2048;                // tensor_dim0_stride[31:0]
    g1[6] = 0;                   // stride0 hi | stride1 lo
    g1[7] = 0;
    const int4v z4 = {0, 0, 0, 0};
#if __clang_major__ >= 23
    const int8v z8 = {0, 0, 0, 0, 0, 0, 0, 0};
    __builtin_amdgcn_tensor_load_to_lds(g0, g1, z4, z4, z8, 0);
#else
    __builtin_amdgcn_tensor_load_to_lds(g0, g1, z4, z4, 0);
#endif
    __builtin_amdgcn_s_wait_tensorcnt(0);         // s_wait_tensorcnt 0
  }
  __syncthreads();

  // A row: K[lr - k] == Krev[(2047 - lr) + k], always in-bounds (padded tail).
  const _Float16* __restrict__ arow =
      Krev + (size_t)h * KREV_LD + (L_ - 1 - (l0 + r));
  const _Float16* __restrict__ bcol = &ldsU[(size_t)r * L_];  // batch column r

  v8f acc = {};
  for (int j0 = 0; j0 < l0 + 16; j0 += 32) {
    __builtin_prefetch(arow + j0 + 128, 0, 1);
    const v16h a = load_tile_rowmajor(arow, j0, half);
    v16h b;
#pragma unroll
    for (int i = 0; i < 8; ++i) b[i] = bcol[j0 + 8 * half + i];
#pragma unroll
    for (int i = 0; i < 8; ++i) b[8 + i] = bcol[j0 + 16 + 8 * half + i];
    acc = __builtin_amdgcn_wmma_f32_16x16x32_f16(false, a, false, b,
                                                 (short)0, acc, false, false);
  }

  const float dh = D[h];
  if (r < B_) {  // N columns 8..15 are zero padding
#pragma unroll
    for (int i = 0; i < 8; ++i) {
      const int l = l0 + i + 8 * half;
      const size_t off = ((size_t)r * L_ + l) * H_ + h;  // out[b, l, h]
      out[off] = acc[i] + dh * u[off];
    }
  }
}

// ---------------------------------------------------------------------------
extern "C" void kernel_launch(void* const* d_in, const int* in_sizes, int n_in,
                              void* d_out, int out_size, void* d_ws, size_t ws_size,
                              hipStream_t stream) {
  (void)in_sizes; (void)n_in; (void)out_size; (void)ws_size;

  const float* u     = (const float*)d_in[0];
  const float* LRe   = (const float*)d_in[1];
  const float* LIm   = (const float*)d_in[2];
  const float* P     = (const float*)d_in[3];
  const float* Bm    = (const float*)d_in[4];
  const float* Cm    = (const float*)d_in[5];
  const float* D     = (const float*)d_in[6];
  const float* lstep = (const float*)d_in[7];
  float* out = (float*)d_out;

  // Workspace (f16): A16 4MB | Wt 16MB | Krev 2.1MB | Ut2 32MB  (~54.1 MB)
  _Float16* A16  = (_Float16*)d_ws;
  _Float16* Wt   = A16  + (size_t)H_ * K2_;
  _Float16* Krev = Wt   + (size_t)L_ * K2_;
  _Float16* Ut2  = Krev + (size_t)H_ * KREV_LD;

  s4_cauchy_kernel<<<H_ * (L_ / 256), 256, 0, stream>>>(LRe, LIm, P, Bm, Cm,
                                                        lstep, A16);
  s4_twiddle_kernel<<<((size_t)L_ * K2_) / 256, 256, 0, stream>>>(Wt);
  s4_transpose_u_kernel<<<((size_t)H_ * L_) / 256, 256, 0, stream>>>(u, Ut2);
  s4_krev_pad_kernel<<<(H_ * 64) / 256, 256, 0, stream>>>(Krev);
  s4_idft_wmma_kernel<<<(32 * 128) / 8, 256, 0, stream>>>(A16, Wt, Krev);
  s4_conv_wmma_kernel<<<(H_ * 128) / 8, 256, 0, stream>>>(Krev, Ut2, u, D, out);
}
